// EncoderLayer_11974368821578
// MI455X (gfx1250) — matrix operand order repaired and checked by reading, not compiled, not run on hardware
//
#include <hip/hip_runtime.h>

typedef __attribute__((ext_vector_type(16))) _Float16 v16h;
typedef __attribute__((ext_vector_type(8)))  float    v8f;

static constexpr int Bx = 4, Nx = 2048, Dx = 256, Hx = 4, HDx = 64, DFFx = 1024;
static constexpr int BN = Bx * Nx;

// ---------------------------------------------------------------- LayerNorm
__global__ void ln_kernel(const float* __restrict__ x, const float* __restrict__ g,
                          const float* __restrict__ bb, float* __restrict__ of32,
                          _Float16* __restrict__ of16) {
  __shared__ float red[256];
  int row = blockIdx.x, tid = threadIdx.x;
  float v = x[(size_t)row * Dx + tid];
  red[tid] = v; __syncthreads();
  for (int s = 128; s > 0; s >>= 1) { if (tid < s) red[tid] += red[tid + s]; __syncthreads(); }
  float mu = red[0] * (1.0f / Dx);
  __syncthreads();
  float d = v - mu;
  red[tid] = d * d; __syncthreads();
  for (int s = 128; s > 0; s >>= 1) { if (tid < s) red[tid] += red[tid + s]; __syncthreads(); }
  float var = red[0] * (1.0f / Dx);
  float xn = d * rsqrtf(var + 1e-5f) * g[tid] + bb[tid];
  if (of32) of32[(size_t)row * Dx + tid] = xn;
  if (of16) of16[(size_t)row * Dx + tid] = (_Float16)xn;
}

// ------------------------------------------- weight transpose + f32 -> f16
__global__ void wconv_kernel(const float* __restrict__ W, _Float16* __restrict__ Wt,
                             int K, int Nout) {
  int idx = blockIdx.x * blockDim.x + threadIdx.x;
  if (idx >= K * Nout) return;
  int k = idx / Nout, n = idx % Nout;
  Wt[(size_t)n * K + k] = (_Float16)W[idx];
}

// --------------------------------------------------------- WMMA GEMM
// C[M,Nout] = A[M,K] @ W[K,Nout] (+bias) (gelu?) (+resid) ; A f16, Bt = W^T f16
// one wave -> 16 rows x 64 cols (4 C tiles)
__global__ void gemm_wmma_kernel(const _Float16* __restrict__ A, const _Float16* __restrict__ Bt,
                                 const float* __restrict__ bias, const float* __restrict__ resid,
                                 float* __restrict__ of32, _Float16* __restrict__ of16,
                                 int M, int K, int Nout, int act) {
  int gw   = (blockIdx.x * blockDim.x + threadIdx.x) >> 5;
  int lane = threadIdx.x & 31;
  int ntm  = M >> 4;
  int mt = gw % ntm, ng = gw / ntm;
  if (ng >= (Nout >> 6)) return;
  int lm = lane & 15;
  int hi = (lane >> 4) & 1;       // lane half
  int hi8 = hi * 8;

  const _Float16* arow = A + (size_t)(mt * 16 + lm) * K;
  v8f vzero = {};
  v8f acc[4];
#pragma unroll
  for (int t = 0; t < 4; t++) acc[t] = vzero;

  for (int kb = 0; kb < K; kb += 32) {
    __builtin_prefetch(arow + kb + 128, 0, 1);
    v16h a;
#pragma unroll
    for (int j = 0; j < 8; j++) {
      a[j]     = arow[kb + hi * 16 + j];     // A layout: K hi*16..hi*16+7
      a[8 + j] = arow[kb + hi * 16 + 8 + j]; //           K hi*16+8..hi*16+15
    }
#pragma unroll
    for (int t = 0; t < 4; t++) {
      const _Float16* brow = Bt + (size_t)(ng * 64 + t * 16 + lm) * K + kb + hi * 16;
      v16h bv;
#pragma unroll
      for (int j = 0; j < 16; j++) bv[j] = brow[j];
      acc[t] = __builtin_amdgcn_wmma_f32_16x16x32_f16(false, a, false, bv,
                                                      (short)0, acc[t], false, false);
    }
  }
#pragma unroll
  for (int t = 0; t < 4; t++) {
    int n = ng * 64 + t * 16 + lm;
    float bvv = bias ? bias[n] : 0.0f;
#pragma unroll
    for (int r = 0; r < 8; r++) {
      int m = mt * 16 + r + hi8;
      float v = acc[t][r] + bvv;
      if (act == 1) v = 0.5f * v * (1.0f + erff(v * 0.70710678118654752f));
      size_t o = (size_t)m * Nout + n;
      if (resid) v += resid[o];
      if (of32) of32[o] = v;
      if (of16) of16[o] = (_Float16)v;
    }
  }
}

// --------------------------------------------------- flash attention w/ bias
// grid: Bx*Hx*16 blocks of 256 threads; wave = one 16-row query tile
// K/V tiles staged with CDNA5 async global->LDS copies (ASYNCcnt).
__global__ void attn_kernel(const _Float16* __restrict__ qf, const _Float16* __restrict__ kf,
                            const _Float16* __restrict__ vf, const float* __restrict__ coords,
                            const unsigned char* __restrict__ pad,
                            const float* __restrict__ semb, const float* __restrict__ temb,
                            _Float16* __restrict__ yf) {
  __shared__ __align__(16) _Float16 Ks[32][64];   // [key][dim]
  __shared__ __align__(16) _Float16 Vs[32][64];   // [key][dim]
  __shared__ float    kcs[32][3];
  __shared__ int      kpadS[32];
  __shared__ _Float16 Ps[8][16][36];              // per-wave P tile (C->A relayout)

  int tid = threadIdx.x;
  int w = tid >> 5, lane = tid & 31;
  int qchunk = blockIdx.x & 15;
  int bh = blockIdx.x >> 4;
  int h = bh & (Hx - 1), b = bh >> 2;
  int qbase = (qchunk * 8 + w) * 16;
  int lm = lane & 15;
  int hi = (lane >> 4) & 1;
  int hi8 = hi * 8;

  size_t qrowg = (size_t)b * Nx + qbase + lm;
  float qt0 = coords[qrowg * 3 + 0];
  float qy0 = coords[qrowg * 3 + 1];
  float qx0 = coords[qrowg * 3 + 2];

  const _Float16* qrow = qf + qrowg * Dx + h * HDx;
  v16h qa0, qa1;
#pragma unroll
  for (int j = 0; j < 8; j++) {
    qa0[j]     = qrow[hi8 + j];        qa0[8 + j] = qrow[16 + hi8 + j];
    qa1[j]     = qrow[32 + hi8 + j];   qa1[8 + j] = qrow[48 + hi8 + j];
  }

  v8f vzero = {};
  v8f oacc[4];
#pragma unroll
  for (int t = 0; t < 4; t++) oacc[t] = vzero;
  float mrun[8], lrun[8];
#pragma unroll
  for (int r = 0; r < 8; r++) { mrun[r] = -1e30f; lrun[r] = 0.0f; }

  // per-thread staging slot: one 16B chunk of K and one of V per key-block
  int s_key = tid >> 3, s_d8 = tid & 7;
  unsigned ldsK = (unsigned)(size_t)&Ks[s_key][s_d8 * 8];
  unsigned ldsV = (unsigned)(size_t)&Vs[s_key][s_d8 * 8];

  for (int kb0 = 0; kb0 < Nx; kb0 += 32) {
    __syncthreads();   // protect LDS tiles still being read from prev iter
    {
      size_t src = ((size_t)b * Nx + kb0 + s_key) * Dx + h * HDx + s_d8 * 8;
      const _Float16* gk = kf + src;
      const _Float16* gv = vf + src;
      // CDNA5 async global->LDS copies (tracked by ASYNCcnt)
      asm volatile("global_load_async_to_lds_b128 %0, %1, off"
                   :: "v"(ldsK), "v"(gk) : "memory");
      asm volatile("global_load_async_to_lds_b128 %0, %1, off"
                   :: "v"(ldsV), "v"(gv) : "memory");
    }
    if (tid < 32) {
      size_t j = (size_t)b * Nx + kb0 + tid;
      kcs[tid][0] = coords[j * 3 + 0];
      kcs[tid][1] = coords[j * 3 + 1];
      kcs[tid][2] = coords[j * 3 + 2];
      kpadS[tid]  = pad[j];
    }
    asm volatile("s_wait_asynccnt 0" ::: "memory");
    __syncthreads();

    // S = Q @ K^T : 2 key tiles x 2 k-steps
    v8f s0 = vzero, s1 = vzero;
#pragma unroll
    for (int kk = 0; kk < 2; kk++) {
      v16h a = kk ? qa1 : qa0;
      int kd = kk * 32 + hi8;
      v16h bk0, bk1;
#pragma unroll
      for (int j = 0; j < 16; j++) { bk0[j] = Ks[lm][kd + j + (j & 8)]; bk1[j] = Ks[16 + lm][kd + j + (j & 8)]; }
      s0 = __builtin_amdgcn_wmma_f32_16x16x32_f16(false, a, false, bk0, (short)0, s0, false, false);
      s1 = __builtin_amdgcn_wmma_f32_16x16x32_f16(false, a, false, bk1, (short)0, s1, false, false);
    }

    // bias + mask + online softmax per row
#pragma unroll
    for (int r = 0; r < 8; r++) {
      int src = r + hi8;
      float qt = __shfl(qt0, src, 32);
      float qy = __shfl(qy0, src, 32);
      float qx = __shfl(qx0, src, 32);
      float sv[2] = { s0[r], s1[r] };
#pragma unroll
      for (int t = 0; t < 2; t++) {
        int jl = t * 16 + lm;
        float dyy = qy - kcs[jl][1];
        float dxx = qx - kcs[jl][2];
        float dist = sqrtf(dyy * dyy + dxx * dxx);
        int sidx = (int)(dist * 0.125f); if (sidx > 31) sidx = 31;
        int dti = (int)(qt - kcs[jl][0]);
        int tix = dti + 16; tix = tix < 0 ? 0 : (tix > 32 ? 32 : tix);
        float bias = semb[sidx * Hx + h] + temb[tix * Hx + h];
        float s = sv[t] * 0.125f + bias;
        if (dist > 256.0f || kpadS[jl]) s = -1e30f;
        sv[t] = s;
      }
      float mx = fmaxf(sv[0], sv[1]);
      for (int off = 8; off; off >>= 1) mx = fmaxf(mx, __shfl_xor(mx, off, 32));
      float mnew  = fmaxf(mrun[r], mx);
      float alpha = __expf(mrun[r] - mnew);
      float p0 = (sv[0] < -1e29f) ? 0.0f : __expf(sv[0] - mnew);
      float p1 = (sv[1] < -1e29f) ? 0.0f : __expf(sv[1] - mnew);
      float rs = p0 + p1;
      for (int off = 8; off; off >>= 1) rs += __shfl_xor(rs, off, 32);
      lrun[r] = lrun[r] * alpha + rs;
      mrun[r] = mnew;
#pragma unroll
      for (int t4 = 0; t4 < 4; t4++) oacc[t4][r] *= alpha;
      Ps[w][r + hi8][lm]      = (_Float16)p0;
      Ps[w][r + hi8][16 + lm] = (_Float16)p1;
    }
    __syncthreads();

    // P (A layout) from LDS, then O += P @ V
    v16h pa;
    {
      const _Float16* pr = &Ps[w][lm][0];
#pragma unroll
      for (int j = 0; j < 8; j++) { pa[j] = pr[hi8 + j]; pa[8 + j] = pr[16 + hi8 + j]; }
    }
#pragma unroll
    for (int t4 = 0; t4 < 4; t4++) {
      v16h bv;
      int nn = t4 * 16 + lm, kbv = hi8;
#pragma unroll
      for (int j = 0; j < 16; j++) bv[j] = Vs[kbv + j + (j & 8)][nn];   // B: K=key striped, N=dim
      oacc[t4] = __builtin_amdgcn_wmma_f32_16x16x32_f16(false, pa, false, bv,
                                                        (short)0, oacc[t4], false, false);
    }
  }

#pragma unroll
  for (int r = 0; r < 8; r++) {
    float inv = lrun[r] > 0.0f ? 1.0f / lrun[r] : 0.0f;
    size_t orow = ((size_t)b * Nx + qbase + r + hi8) * Dx + h * HDx;
#pragma unroll
    for (int t4 = 0; t4 < 4; t4++)
      yf[orow + t4 * 16 + lm] = (_Float16)(oacc[t4][r] * inv);
  }
}

// -------------------------------------------------------------------- launch
extern "C" void kernel_launch(void* const* d_in, const int* in_sizes, int n_in,
                              void* d_out, int out_size, void* d_ws, size_t ws_size,
                              hipStream_t stream) {
  const float* x      = (const float*)d_in[0];
  const float* coords = (const float*)d_in[1];
  const unsigned char* pmask = (const unsigned char*)d_in[2];
  const float* n1g = (const float*)d_in[3];
  const float* n1b = (const float*)d_in[4];
  const float* Wq = (const float*)d_in[5];  const float* bq = (const float*)d_in[6];
  const float* Wk = (const float*)d_in[7];  const float* bk = (const float*)d_in[8];
  const float* Wv = (const float*)d_in[9];  const float* bv = (const float*)d_in[10];
  const float* Wo = (const float*)d_in[11]; const float* bo = (const float*)d_in[12];
  const float* semb = (const float*)d_in[13];
  const float* temb = (const float*)d_in[14];
  const float* n2g = (const float*)d_in[15];
  const float* n2b = (const float*)d_in[16];
  const float* W1 = (const float*)d_in[17]; const float* b1 = (const float*)d_in[18];
  const float* W2 = (const float*)d_in[19]; const float* b2 = (const float*)d_in[20];
  float* out = (float*)d_out;

  char* wsp = (char*)d_ws;
  auto alloc = [&](size_t bytes) -> char* {
    char* p = wsp; wsp += (bytes + 255) & ~(size_t)255; return p;
  };
  _Float16* WqT = (_Float16*)alloc((size_t)Dx * Dx * 2);
  _Float16* WkT = (_Float16*)alloc((size_t)Dx * Dx * 2);
  _Float16* WvT = (_Float16*)alloc((size_t)Dx * Dx * 2);
  _Float16* WoT = (_Float16*)alloc((size_t)Dx * Dx * 2);
  _Float16* W1T = (_Float16*)alloc((size_t)Dx * DFFx * 2);
  _Float16* W2T = (_Float16*)alloc((size_t)DFFx * Dx * 2);
  _Float16* xn16 = (_Float16*)alloc((size_t)BN * Dx * 2);
  float*    xn32 = (float*)   alloc((size_t)BN * Dx * 4);
  _Float16* q16  = (_Float16*)alloc((size_t)BN * Dx * 2);
  _Float16* k16  = (_Float16*)alloc((size_t)BN * Dx * 2);
  _Float16* v16  = (_Float16*)alloc((size_t)BN * Dx * 2);
  _Float16* y16  = (_Float16*)alloc((size_t)BN * Dx * 2);
  float*    h32  = (float*)   alloc((size_t)BN * Dx * 4);
  _Float16* hn16 = (_Float16*)alloc((size_t)BN * Dx * 2);
  _Float16* m116 = (_Float16*)alloc((size_t)BN * DFFx * 2);

  // weight prep
  wconv_kernel<<<(Dx * Dx + 255) / 256, 256, 0, stream>>>(Wq, WqT, Dx, Dx);
  wconv_kernel<<<(Dx * Dx + 255) / 256, 256, 0, stream>>>(Wk, WkT, Dx, Dx);
  wconv_kernel<<<(Dx * Dx + 255) / 256, 256, 0, stream>>>(Wv, WvT, Dx, Dx);
  wconv_kernel<<<(Dx * Dx + 255) / 256, 256, 0, stream>>>(Wo, WoT, Dx, Dx);
  wconv_kernel<<<(Dx * DFFx + 255) / 256, 256, 0, stream>>>(W1, W1T, Dx, DFFx);
  wconv_kernel<<<(DFFx * Dx + 255) / 256, 256, 0, stream>>>(W2, W2T, DFFx, Dx);

  // LN1
  ln_kernel<<<BN, 256, 0, stream>>>(x, n1g, n1b, xn32, xn16);

  // QKV projections
  int gblk_d = (BN / 16) * (Dx / 64) / 8;      // 256 blocks
  gemm_wmma_kernel<<<gblk_d, 256, 0, stream>>>(xn16, WqT, bq, nullptr, nullptr, q16, BN, Dx, Dx, 0);
  gemm_wmma_kernel<<<gblk_d, 256, 0, stream>>>(xn16, WkT, bk, nullptr, nullptr, k16, BN, Dx, Dx, 0);
  gemm_wmma_kernel<<<gblk_d, 256, 0, stream>>>(xn16, WvT, bv, nullptr, nullptr, v16, BN, Dx, Dx, 0);

  // attention
  attn_kernel<<<Bx * Hx * 16, 256, 0, stream>>>(q16, k16, v16, coords, pmask, semb, temb, y16);

  // output projection + residual (h = xn + y@Wo+bo)
  gemm_wmma_kernel<<<gblk_d, 256, 0, stream>>>(y16, WoT, bo, xn32, h32, nullptr, BN, Dx, Dx, 0);

  // LN2
  ln_kernel<<<BN, 256, 0, stream>>>(h32, n2g, n2b, nullptr, hn16);

  // MLP
  int gblk_ff = (BN / 16) * (DFFx / 64) / 8;   // 1024 blocks
  gemm_wmma_kernel<<<gblk_ff, 256, 0, stream>>>(hn16, W1T, b1, nullptr, nullptr, m116, BN, Dx, DFFx, 1);
  gemm_wmma_kernel<<<gblk_d, 256, 0, stream>>>(m116, W2T, b2, h32, out, nullptr, BN, DFFx, Dx, 0);
}
